// Block1SyntaxEngine_85959475462663
// MI455X (gfx1250) — matrix operand order repaired and checked by reading, not compiled, not run on hardware
//
#include <hip/hip_runtime.h>
#include <math.h>

#define BB   2
#define TT   1024
#define DD   1024
#define HH   16
#define LL   6
#define DHV  64
#define DFF_ 2048
#define BT_  (BB*TT)

typedef __attribute__((ext_vector_type(16))) __bf16       v16bf;
typedef __attribute__((ext_vector_type(8)))  float        v8f;
typedef __attribute__((ext_vector_type(4)))  unsigned int v4u;
typedef __attribute__((ext_vector_type(4)))  int          v4i;

union Frag { v4u u[2]; v16bf v; };

#if __has_builtin(__builtin_amdgcn_global_load_async_to_lds_b128)
#define HAVE_ASYNC_LDS 1
#else
#define HAVE_ASYNC_LDS 0
#endif

__device__ __forceinline__ unsigned short f2bf(float f) {
  unsigned int u = __float_as_uint(f);
  u += 0x7FFFu + ((u >> 16) & 1u);   // round-to-nearest-even
  return (unsigned short)(u >> 16);
}

__device__ __forceinline__ v8f vzero8() {
  v8f z;
#pragma unroll
  for (int i = 0; i < 8; i++) z[i] = 0.f;
  return z;
}

__device__ __forceinline__ v8f wmma_bf16(const Frag& a, const Frag& b, v8f c) {
  return __builtin_amdgcn_wmma_f32_16x16x32_bf16(false, a.v, false, b.v,
                                                 (short)0, c, false, false);
}

// 16-byte global -> LDS copy: async on CDNA5 (ASYNCcnt), sync fallback.
__device__ __forceinline__ void copy16_g2l(const unsigned short* g,
                                           unsigned short* l) {
#if HAVE_ASYNC_LDS
  __builtin_amdgcn_global_load_async_to_lds_b128(
      (__attribute__((address_space(1))) v4i*)g,
      (__attribute__((address_space(3))) v4i*)l, 0, 0);
#else
  *(v4u*)l = *(const v4u*)g;
#endif
}

template <int N>
__device__ __forceinline__ void wait_async() {
#if HAVE_ASYNC_LDS
#if __has_builtin(__builtin_amdgcn_s_wait_asynccnt)
  __builtin_amdgcn_s_wait_asynccnt(N);
#else
  asm volatile("s_wait_asynccnt %0" ::"i"(N) : "memory");
#endif
#endif
}

// ---------------------------------------------------------------------------
// Embedding: x[bt][d] = token_emb[ids[bt]][d] + pos_emb[t][d]
// ---------------------------------------------------------------------------
__global__ __launch_bounds__(256) void embed_kernel(const int* __restrict__ ids,
                                                    const float* __restrict__ tok,
                                                    const float* __restrict__ pos,
                                                    float* __restrict__ x) {
  int bt = blockIdx.x;
  int t  = bt & (TT - 1);
  int id = ids[bt];
  const float4* te = (const float4*)(tok + (size_t)id * DD);
  const float4* pe = (const float4*)(pos + (size_t)t * DD);
  float4* xo = (float4*)(x + (size_t)bt * DD);
  int i = threadIdx.x;
  float4 a = te[i], b = pe[i];
  xo[i] = make_float4(a.x + b.x, a.y + b.y, a.z + b.z, a.w + b.w);
}

// ---------------------------------------------------------------------------
// LayerNorm over D=1024. One block per row, 256 threads * 4 elems.
// ---------------------------------------------------------------------------
template <bool BF16OUT>
__global__ __launch_bounds__(256) void ln_kernel(const float* __restrict__ x,
                                                 const float* __restrict__ gam,
                                                 const float* __restrict__ bet,
                                                 void* __restrict__ out) {
  __shared__ float ssum[8], ssq[8];
  int row = blockIdx.x;
  int tid = threadIdx.x, wid = tid >> 5, lane = tid & 31;
  const float4* xr = (const float4*)(x + (size_t)row * DD);
  float4 v = xr[tid];
  float sum = v.x + v.y + v.z + v.w;
  float sq  = v.x * v.x + v.y * v.y + v.z * v.z + v.w * v.w;
#pragma unroll
  for (int off = 1; off < 32; off <<= 1) {
    sum += __shfl_xor(sum, off, 32);
    sq  += __shfl_xor(sq,  off, 32);
  }
  if (lane == 0) { ssum[wid] = sum; ssq[wid] = sq; }
  __syncthreads();
  if (tid == 0) {
    float a = 0.f, b = 0.f;
#pragma unroll
    for (int i = 0; i < 8; i++) { a += ssum[i]; b += ssq[i]; }
    ssum[0] = a; ssq[0] = b;
  }
  __syncthreads();
  float mean = ssum[0] * (1.0f / DD);
  float var  = ssq[0] * (1.0f / DD) - mean * mean;
  float rs   = rsqrtf(var + 1e-5f);
  float4 g  = ((const float4*)gam)[tid];
  float4 b4 = ((const float4*)bet)[tid];
  float o0 = (v.x - mean) * rs * g.x + b4.x;
  float o1 = (v.y - mean) * rs * g.y + b4.y;
  float o2 = (v.z - mean) * rs * g.z + b4.z;
  float o3 = (v.w - mean) * rs * g.w + b4.w;
  if (BF16OUT) {
    unsigned short* o = (unsigned short*)out + (size_t)row * DD + tid * 4;
    o[0] = f2bf(o0); o[1] = f2bf(o1); o[2] = f2bf(o2); o[3] = f2bf(o3);
  } else {
    ((float4*)out)[(size_t)row * (DD / 4) + tid] = make_float4(o0, o1, o2, o3);
  }
}

// ---------------------------------------------------------------------------
// fp32 [K][N] -> bf16 [N][K] transpose (weights into WMMA-B-friendly layout)
// ---------------------------------------------------------------------------
__global__ void transpose_bf16_kernel(const float* __restrict__ in,
                                      unsigned short* __restrict__ out,
                                      int K, int N) {
  __shared__ float tile[32][33];
  int n0 = blockIdx.x * 32, k0 = blockIdx.y * 32;
  int tx = threadIdx.x, ty = threadIdx.y;
#pragma unroll
  for (int i = 0; i < 4; i++)
    tile[ty + i * 8][tx] = in[(size_t)(k0 + ty + i * 8) * N + n0 + tx];
  __syncthreads();
#pragma unroll
  for (int i = 0; i < 4; i++)
    out[(size_t)(n0 + ty + i * 8) * K + k0 + tx] = f2bf(tile[tx][ty + i * 8]);
}

// ---------------------------------------------------------------------------
// WMMA bf16 GEMM: C[M][N] = A[M][K] @ W^T[N][K], fused epilogues.
// 256 threads = 8 waves; block tile 128x128; wave tile 32x64 (2x4 WMMA tiles).
// K-tiles of 32 staged in double-buffered LDS via async global->LDS copies.
// ---------------------------------------------------------------------------
enum { EPI_RES = 0, EPI_GELU = 1, EPI_QKV = 2 };

template <int EPI>
__global__ __launch_bounds__(256) void gemm_kernel(
    const unsigned short* __restrict__ A, const unsigned short* __restrict__ W,
    int K, int N, float* __restrict__ resid, unsigned short* __restrict__ outb,
    unsigned short* __restrict__ Qp, unsigned short* __restrict__ Kp,
    unsigned short* __restrict__ Vp, float qscale) {
  constexpr int BK = 32;
  constexpr int SK = 40;  // padded row stride (80 B: 16-B aligned, bank-spread)
  __shared__ __align__(16) unsigned short smem[2][2][128 * SK];  // 40 KB

  int tid = threadIdx.x;
  int wid = tid >> 5, lane = tid & 31;
  int half = lane >> 4, ln = lane & 15;
  int wm = wid >> 1, wn = wid & 1;
  int rowTile = blockIdx.y * 128;
  int colTile = blockIdx.x * 128;

  // per-thread staging chunks: 512 b128 chunks per 128x32 tile -> 2/thread/mat
  auto stage = [&](int buf, int kb) {
#pragma unroll
    for (int i = 0; i < 2; i++) {
      int c = tid + 256 * i;
      int row = c >> 2, kc = (c & 3) * 8;
      copy16_g2l(A + (size_t)(rowTile + row) * K + kb + kc,
                 &smem[buf][0][row * SK + kc]);
      copy16_g2l(W + (size_t)(colTile + row) * K + kb + kc,
                 &smem[buf][1][row * SK + kc]);
    }
  };

  v8f c[2][4];
#pragma unroll
  for (int m = 0; m < 2; m++)
#pragma unroll
    for (int n = 0; n < 4; n++) c[m][n] = vzero8();

  stage(0, 0);
  int ntiles = K / BK;
  for (int t = 0; t < ntiles; t++) {
    int buf = t & 1;
    if (t + 1 < ntiles) {
      stage(buf ^ 1, (t + 1) * BK);
      wait_async<4>();  // in-order: current tile's 4 async copies landed
    } else {
      wait_async<0>();
    }
    __syncthreads();
    const unsigned short* As = &smem[buf][0][0];
    const unsigned short* Bs = &smem[buf][1][0];
    Frag a0, a1;
    const unsigned short* ap0 = As + (wm * 32 + ln) * SK;
    const unsigned short* ap1 = As + (wm * 32 + 16 + ln) * SK;
    a0.u[0] = *(const v4u*)(ap0 + half * 8);
    a0.u[1] = *(const v4u*)(ap0 + 16 + half * 8);
    a1.u[0] = *(const v4u*)(ap1 + half * 8);
    a1.u[1] = *(const v4u*)(ap1 + 16 + half * 8);
#pragma unroll
    for (int n = 0; n < 4; n++) {
      Frag b;
      const unsigned short* bp = Bs + (wn * 64 + n * 16 + ln) * SK;
      b.u[0] = *(const v4u*)(bp + half * 8);
      b.u[1] = *(const v4u*)(bp + half * 8 + 16);
      c[0][n] = wmma_bf16(a0, b, c[0][n]);
      c[1][n] = wmma_bf16(a1, b, c[1][n]);
    }
    __syncthreads();
  }

#pragma unroll
  for (int m = 0; m < 2; m++)
#pragma unroll
    for (int n = 0; n < 4; n++)
#pragma unroll
      for (int r = 0; r < 8; r++) {
        int row = rowTile + wm * 32 + m * 16 + r + half * 8;
        int col = colTile + wn * 64 + n * 16 + ln;
        float val = c[m][n][r];
        if (EPI == EPI_RES) {
          size_t idx = (size_t)row * N + col;
          resid[idx] = resid[idx] + val;
        } else if (EPI == EPI_GELU) {
          float g = 0.5f * val * (1.f + erff(val * 0.70710678118f));
          outb[(size_t)row * N + col] = f2bf(g);
        } else {  // EPI_QKV: split into Q (scaled), K (row-major), V^T
          int which = col >> 10;  // /D
          int cd = col & 1023;
          int hh = cd >> 6;       // /DH
          int dh = cd & 63;
          int b = row >> 10;      // /T
          int t = row & 1023;
          if (which == 0)
            Qp[((size_t)(b * HH + hh) * TT + t) * DHV + dh] = f2bf(val * qscale);
          else if (which == 1)
            Kp[((size_t)(b * HH + hh) * TT + t) * DHV + dh] = f2bf(val);
          else
            Vp[((size_t)(b * HH + hh) * DHV + dh) * TT + t] = f2bf(val);
        }
      }
}

// ---------------------------------------------------------------------------
// Flash attention, causal. One wave per 16-row query block per (b,h).
// Key blocks of 32: S via 4 WMMAs, online softmax, P->LDS transpose, AV via
// 4 WMMAs into 16x64 fp32 accumulator. No block barriers (divergent waves).
// ---------------------------------------------------------------------------
__global__ __launch_bounds__(256) void attn_kernel(
    const unsigned short* __restrict__ Q, const unsigned short* __restrict__ Kb,
    const unsigned short* __restrict__ Vt, unsigned short* __restrict__ O) {
  __shared__ __align__(16) unsigned short pbuf[8][16 * 32];
  int tid = threadIdx.x, wid = tid >> 5, lane = tid & 31;
  int half = lane >> 4, ln = lane & 15;
  int gq = blockIdx.x * 8 + wid;
  int qb = gq & 63;  // T/16 = 64
  int tmpv = gq >> 6;
  int h = tmpv & 15;
  int b = tmpv >> 4;
  int i0 = qb * 16;

  const unsigned short* qrow = Q + ((size_t)(b * HH + h) * TT + i0 + ln) * DHV;
  Frag qf[2];
#pragma unroll
  for (int kk = 0; kk < 2; kk++) {
    qf[kk].u[0] = *(const v4u*)(qrow + kk * 32 + half * 8);
    qf[kk].u[1] = *(const v4u*)(qrow + kk * 32 + 16 + half * 8);
  }

  float m_[8], l_[8];
  v8f o[4];
#pragma unroll
  for (int r = 0; r < 8; r++) { m_[r] = -1e30f; l_[r] = 0.f; }
#pragma unroll
  for (int n = 0; n < 4; n++) o[n] = vzero8();

  const unsigned short* kbase = Kb + (size_t)(b * HH + h) * TT * DHV;
  const unsigned short* vbase = Vt + (size_t)(b * HH + h) * DHV * TT;
  unsigned short* pw = &pbuf[wid][0];

  int nblocks = (i0 + 47) >> 5;  // key blocks of 32 covering keys <= i0+15
  for (int blk = 0; blk < nblocks; blk++) {
    int jb = blk * 32;
    v8f s0 = vzero8(), s1 = vzero8();
#pragma unroll
    for (int kk = 0; kk < 2; kk++) {
      Frag kf0, kf1;
      const unsigned short* kr0 = kbase + (size_t)(jb + ln) * DHV + kk * 32 + half * 8;
      const unsigned short* kr1 = kbase + (size_t)(jb + 16 + ln) * DHV + kk * 32 + half * 8;
      kf0.u[0] = *(const v4u*)kr0;  kf0.u[1] = *(const v4u*)(kr0 + 16);
      kf1.u[0] = *(const v4u*)kr1;  kf1.u[1] = *(const v4u*)(kr1 + 16);
      s0 = wmma_bf16(qf[kk], kf0, s0);
      s1 = wmma_bf16(qf[kk], kf1, s1);
    }
    float p0[8], p1[8];
#pragma unroll
    for (int r = 0; r < 8; r++) {
      int row = i0 + r + half * 8;
      int c0 = jb + ln, c1 = jb + 16 + ln;
      float v0 = (c0 <= row) ? s0[r] : -1e30f;
      float v1 = (c1 <= row) ? s1[r] : -1e30f;
      float mx = fmaxf(v0, v1);
#pragma unroll
      for (int off = 1; off < 16; off <<= 1) mx = fmaxf(mx, __shfl_xor(mx, off, 32));
      float mnew = fmaxf(m_[r], mx);
      float scale = __expf(m_[r] - mnew);
      float e0 = __expf(v0 - mnew), e1 = __expf(v1 - mnew);
      float sm = e0 + e1;
#pragma unroll
      for (int off = 1; off < 16; off <<= 1) sm += __shfl_xor(sm, off, 32);
      l_[r] = l_[r] * scale + sm;
      m_[r] = mnew;
#pragma unroll
      for (int n = 0; n < 4; n++) o[n][r] *= scale;
      p0[r] = e0; p1[r] = e1;
    }
#pragma unroll
    for (int r = 0; r < 8; r++) {
      pw[(r + half * 8) * 32 + ln]      = f2bf(p0[r]);
      pw[(r + half * 8) * 32 + 16 + ln] = f2bf(p1[r]);
    }
    asm volatile("s_wait_dscnt 0" ::: "memory");  // per-wave LDS RAW fence
    Frag pf;
    pf.u[0] = *(const v4u*)(pw + ln * 32 + half * 8);
    pf.u[1] = *(const v4u*)(pw + ln * 32 + 16 + half * 8);
#pragma unroll
    for (int n = 0; n < 4; n++) {
      Frag vf;
      const unsigned short* vr = vbase + (size_t)(n * 16 + ln) * TT + jb + half * 8;
      vf.u[0] = *(const v4u*)vr;
      vf.u[1] = *(const v4u*)(vr + 16);
      o[n] = wmma_bf16(pf, vf, o[n]);
    }
    asm volatile("" ::: "memory");
  }
#pragma unroll
  for (int n = 0; n < 4; n++)
#pragma unroll
    for (int r = 0; r < 8; r++) {
      float val = o[n][r] / l_[r];
      O[(size_t)(b * TT + i0 + r + half * 8) * DD + h * DHV + n * 16 + ln] = f2bf(val);
    }
}

// ---------------------------------------------------------------------------
extern "C" void kernel_launch(void* const* d_in, const int* in_sizes, int n_in,
                              void* d_out, int out_size, void* d_ws, size_t ws_size,
                              hipStream_t stream) {
  const int*   ids  = (const int*)d_in[0];
  const float* tok  = (const float*)d_in[1];
  const float* pos  = (const float*)d_in[2];
  const float* ln1s = (const float*)d_in[3];
  const float* ln1b = (const float*)d_in[4];
  const float* Wqkv = (const float*)d_in[5];
  const float* Wout = (const float*)d_in[6];
  const float* ln2s = (const float*)d_in[7];
  const float* ln2b = (const float*)d_in[8];
  const float* W1   = (const float*)d_in[9];
  const float* W2   = (const float*)d_in[10];
  const float* lnfs = (const float*)d_in[11];
  const float* lnfb = (const float*)d_in[12];

  char* ws = (char*)d_ws;
  size_t off = 0;
  auto take = [&](size_t bytes) {
    char* p = ws + off;
    off += (bytes + 255) & ~(size_t)255;
    return p;
  };
  unsigned short* wqkvT = (unsigned short*)take((size_t)3 * DD * DD * 2);
  unsigned short* woutT = (unsigned short*)take((size_t)DD * DD * 2);
  unsigned short* w1T   = (unsigned short*)take((size_t)DFF_ * DD * 2);
  unsigned short* w2T   = (unsigned short*)take((size_t)DD * DFF_ * 2);
  float*          x     = (float*)take((size_t)BT_ * DD * 4);
  unsigned short* hbuf  = (unsigned short*)take((size_t)BT_ * DD * 2);
  unsigned short* qbuf  = (unsigned short*)take((size_t)BB * HH * TT * DHV * 2);
  unsigned short* kbuf  = (unsigned short*)take((size_t)BB * HH * TT * DHV * 2);
  unsigned short* vtbuf = (unsigned short*)take((size_t)BB * HH * TT * DHV * 2);
  unsigned short* obuf  = (unsigned short*)take((size_t)BT_ * DD * 2);
  unsigned short* gbuf  = (unsigned short*)take((size_t)BT_ * DFF_ * 2);
  (void)ws_size; (void)in_sizes; (void)n_in; (void)out_size;

  embed_kernel<<<BT_, 256, 0, stream>>>(ids, tok, pos, x);

  dim3 tb(32, 8);
  for (int l = 0; l < LL; l++) {
    transpose_bf16_kernel<<<dim3(3 * DD / 32, DD / 32), tb, 0, stream>>>(
        Wqkv + (size_t)l * DD * 3 * DD, wqkvT, DD, 3 * DD);
    transpose_bf16_kernel<<<dim3(DD / 32, DD / 32), tb, 0, stream>>>(
        Wout + (size_t)l * DD * DD, woutT, DD, DD);
    transpose_bf16_kernel<<<dim3(DFF_ / 32, DD / 32), tb, 0, stream>>>(
        W1 + (size_t)l * DD * DFF_, w1T, DD, DFF_);
    transpose_bf16_kernel<<<dim3(DD / 32, DFF_ / 32), tb, 0, stream>>>(
        W2 + (size_t)l * DFF_ * DD, w2T, DFF_, DD);

    ln_kernel<true><<<BT_, 256, 0, stream>>>(x, ln1s + l * DD, ln1b + l * DD, hbuf);
    gemm_kernel<EPI_QKV><<<dim3(3 * DD / 128, BT_ / 128), 256, 0, stream>>>(
        hbuf, wqkvT, DD, 3 * DD, nullptr, nullptr, qbuf, kbuf, vtbuf, 0.125f);
    attn_kernel<<<(BB * HH * (TT / 16)) / 8, 256, 0, stream>>>(qbuf, kbuf, vtbuf, obuf);
    gemm_kernel<EPI_RES><<<dim3(DD / 128, BT_ / 128), 256, 0, stream>>>(
        obuf, woutT, DD, DD, x, nullptr, nullptr, nullptr, nullptr, 0.f);
    ln_kernel<true><<<BT_, 256, 0, stream>>>(x, ln2s + l * DD, ln2b + l * DD, hbuf);
    gemm_kernel<EPI_GELU><<<dim3(DFF_ / 128, BT_ / 128), 256, 0, stream>>>(
        hbuf, w1T, DD, DFF_, nullptr, gbuf, nullptr, nullptr, nullptr, 0.f);
    gemm_kernel<EPI_RES><<<dim3(DD / 128, BT_ / 128), 256, 0, stream>>>(
        gbuf, w2T, DFF_, DD, x, nullptr, nullptr, nullptr, nullptr, 0.f);
  }
  ln_kernel<false><<<BT_, 256, 0, stream>>>(x, lnfs, lnfb, d_out);
}
